// LSTMModel_34961033789514
// MI455X (gfx1250) — matrix operand order repaired and checked by reading, not compiled, not run on hardware
//
#include <hip/hip_runtime.h>
#include <math.h>

typedef _Float16 h16;
typedef __attribute__((ext_vector_type(16))) _Float16 v16h;
typedef __attribute__((ext_vector_type(8)))  _Float16 v8h;
typedef __attribute__((ext_vector_type(8)))  float    v8f;

#define B_   512
#define T_   512
#define IN_  32
#define H1_  128
#define H2_  64
#define H3_  32
#define NTHREADS 256
#define RPB  16   // batch rows per block (WMMA M)

// ---- LDS layout (element counts; all h16 sections first, then f32 biases) ----
constexpr int HC_WIH1 = 4 * H1_ * IN_;   // 16384
constexpr int HC_WHH1 = 4 * H1_ * H1_;   // 65536
constexpr int HC_WIH2 = 4 * H2_ * H1_;   // 32768
constexpr int HC_WHH2 = 4 * H2_ * H2_;   // 16384
constexpr int HC_WIH3 = 4 * H3_ * H2_;   // 8192
constexpr int HC_WHH3 = 4 * H3_ * H3_;   // 4096
constexpr int HC_XT   = RPB * IN_;       // 512
constexpr int HC_H1   = RPB * H1_;       // 2048 (x2 ping-pong)
constexpr int HC_H2   = RPB * H2_;       // 1024 (x2)
constexpr int HC_H3   = RPB * H3_;       // 512  (x2)

constexpr int O_WIH1 = 0;
constexpr int O_WHH1 = O_WIH1 + HC_WIH1;
constexpr int O_WIH2 = O_WHH1 + HC_WHH1;
constexpr int O_WHH2 = O_WIH2 + HC_WIH2;
constexpr int O_WIH3 = O_WHH2 + HC_WHH2;
constexpr int O_WHH3 = O_WIH3 + HC_WIH3;
constexpr int O_XT   = O_WHH3 + HC_WHH3;
constexpr int O_H1   = O_XT   + HC_XT;           // two buffers back-to-back
constexpr int O_H2   = O_H1   + 2 * HC_H1;
constexpr int O_H3   = O_H2   + 2 * HC_H2;
constexpr int HALF_TOTAL = O_H3 + 2 * HC_H3;

constexpr int FB1 = 4 * H1_;   // 512
constexpr int FB2 = 4 * H2_;   // 256
constexpr int FB3 = 4 * H3_;   // 128
constexpr size_t LDS_BYTES = (size_t)HALF_TOTAL * 2 + (size_t)(FB1 + FB2 + FB3) * 4; // 305664 B < 320KB

__device__ __forceinline__ float fast_sigmoid(float x) {
  return 1.0f / (1.0f + __expf(-x));
}
__device__ __forceinline__ float fast_tanh(float x) {
  x = fminf(fmaxf(x, -15.0f), 15.0f);
  float e = __expf(2.0f * x);
  return (e - 1.0f) / (e + 1.0f);
}

// A operand: 16xK tile (row-major in LDS, row stride ldk), K-slice [kofs, kofs+32).
// ISA 16-bit A layout: lanes 0-15 hold K=kofs+[0..8)+[16..24), lanes 16-31 K=kofs+[8..16)+[24..32).
template<bool RELU>
__device__ __forceinline__ v16h load_A(const h16* __restrict__ S, int ldk, int kofs, int lane) {
  const int m  = lane & 15;
  const int kb = kofs + ((lane >> 4) << 3);
  const h16* p = S + m * ldk + kb;          // 16B aligned by construction
  v8h lo = *(const v8h*)p;
  v8h hi = *(const v8h*)(p + 16);
  v16h a;
#pragma unroll
  for (int j = 0; j < 8; ++j) {
    h16 x0 = lo[j], x1 = hi[j];
    if (RELU) {
      x0 = (x0 > (h16)0.f) ? x0 : (h16)0.f;
      x1 = (x1 > (h16)0.f) ? x1 : (h16)0.f;
    }
    a[j] = x0; a[8 + j] = x1;
  }
  return a;
}

// B operand: K x 16 slice of W^T where W is [4H x ldk] row-major in LDS.
// B[k][n] = W[n0+n][kofs+k]. Same k order as A: lanes 0-15 hold K=kofs+[0..8)+[16..24),
// lanes 16-31 K=kofs+[8..16)+[24..32), column n = lane&15. Two 16B vector loads.
__device__ __forceinline__ v16h load_B(const h16* __restrict__ W, int ldk, int n0, int kofs, int lane) {
  const int n  = n0 + (lane & 15);
  const int kb = kofs + ((lane >> 4) << 3);
  const h16* p = W + n * ldk + kb;          // 16B aligned by construction
  v8h lo = *(const v8h*)p;
  v8h hi = *(const v8h*)(p + 16);
  return __builtin_shufflevector(lo, hi, 0,1,2,3,4,5,6,7,8,9,10,11,12,13,14,15);
}

template<int KIN, int H>
struct LSTMTile {
  static constexpr int NKI = KIN / 32;
  static constexpr int NKH = H / 32;
  static constexpr int NK  = NKI + NKH;
  v8f c;

  __device__ __forceinline__ void init() {
#pragma unroll
    for (int j = 0; j < 8; ++j) c[j] = 0.0f;
  }

  // Fused load + 4-gate interleaved WMMA (double-buffered B) + cell update.
  // Reads h(t-1) from Hold, writes h(t) to Hnew (ping-pong: no intra-phase barrier needed).
  template<bool RELU_IN>
  __device__ __forceinline__ void step(const h16* __restrict__ A,      // 16 x KIN input at t
                                       const h16* __restrict__ Hold,   // 16 x H  h(t-1)
                                       h16* __restrict__ Hnew,         // 16 x H  h(t)
                                       const h16* __restrict__ Wih,    // 4H x KIN
                                       const h16* __restrict__ Whh,    // 4H x H
                                       const float* __restrict__ bias, // 4H (folded bih+bhh)
                                       int n0, int lane) {
    // ---- A operands for all K-slices ----
    v16h a[NK];
#pragma unroll
    for (int i = 0; i < NKI; ++i) a[i]       = load_A<RELU_IN>(A, KIN, 32 * i, lane);
#pragma unroll
    for (int i = 0; i < NKH; ++i) a[NKI + i] = load_A<false>(Hold, H, 32 * i, lane);

    v8f acc[4];
#pragma unroll
    for (int q = 0; q < 4; ++q)
#pragma unroll
      for (int j = 0; j < 8; ++j) acc[q][j] = 0.0f;

    // ---- double-buffered B operands: 4 gate tiles per K-slice ----
    auto loadBs = [&](int s, v16h* dst) {
#pragma unroll
      for (int q = 0; q < 4; ++q) {
        if (s < NKI) dst[q] = load_B(Wih, KIN, q * H + n0, 32 * s, lane);
        else         dst[q] = load_B(Whh, H,   q * H + n0, 32 * (s - NKI), lane);
      }
    };

    v16h b[2][4];
    loadBs(0, b[0]);
#pragma unroll
    for (int s = 0; s < NK; ++s) {
      if (s + 1 < NK) loadBs(s + 1, b[(s + 1) & 1]);   // prefetch next slice while computing
#pragma unroll
      for (int q = 0; q < 4; ++q)
        acc[q] = __builtin_amdgcn_wmma_f32_16x16x32_f16(
            false, a[s], false, b[s & 1][q], (short)0, acc[q], false, false);
    }

    // ---- elementwise LSTM cell update (bias folded here) ----
    const int n  = n0 + (lane & 15);
    const int mo = (lane >> 4) << 3;
    const float bi = bias[0 * H + n];
    const float bf = bias[1 * H + n];
    const float bg = bias[2 * H + n];
    const float bo = bias[3 * H + n];
#pragma unroll
    for (int j = 0; j < 8; ++j) {
      const float iv = fast_sigmoid(acc[0][j] + bi);
      const float fv = fast_sigmoid(acc[1][j] + bf);
      const float gv = fast_tanh   (acc[2][j] + bg);
      const float ov = fast_sigmoid(acc[3][j] + bo);
      const float cv = fv * c[j] + iv * gv;
      c[j] = cv;
      const float hv = ov * fast_tanh(cv);
      Hnew[(mo + j) * H + n] = (h16)hv;
    }
  }
};

__launch_bounds__(NTHREADS, 1)
__global__ void lstm3_fused_kernel(
    const float* __restrict__ x,
    const float* __restrict__ Wih1, const float* __restrict__ Whh1,
    const float* __restrict__ bih1, const float* __restrict__ bhh1,
    const float* __restrict__ Wih2, const float* __restrict__ Whh2,
    const float* __restrict__ bih2, const float* __restrict__ bhh2,
    const float* __restrict__ Wih3, const float* __restrict__ Whh3,
    const float* __restrict__ bih3, const float* __restrict__ bhh3,
    const float* __restrict__ Wd,   const float* __restrict__ bd,
    float* __restrict__ out)
{
  extern __shared__ char smem[];
  h16*   hb    = (h16*)smem;
  h16*   sWih1 = hb + O_WIH1;
  h16*   sWhh1 = hb + O_WHH1;
  h16*   sWih2 = hb + O_WIH2;
  h16*   sWhh2 = hb + O_WHH2;
  h16*   sWih3 = hb + O_WIH3;
  h16*   sWhh3 = hb + O_WHH3;
  h16*   sXt   = hb + O_XT;
  h16*   sH1   = hb + O_H1;   // 2 buffers of HC_H1
  h16*   sH2   = hb + O_H2;   // 2 buffers of HC_H2
  h16*   sH3   = hb + O_H3;   // 2 buffers of HC_H3
  float* sB1   = (float*)(hb + HALF_TOTAL);
  float* sB2   = sB1 + FB1;
  float* sB3   = sB2 + FB2;

  const int tid  = threadIdx.x;
  const int lane = tid & 31;
  const int wave = tid >> 5;
  const int row0 = blockIdx.x * RPB;

  // ---- Stage all weights into LDS as f16, fold biases, zero h(-1) (buffer 0) ----
  for (int i = tid; i < HC_WIH1; i += NTHREADS) sWih1[i] = (h16)Wih1[i];
  for (int i = tid; i < HC_WHH1; i += NTHREADS) sWhh1[i] = (h16)Whh1[i];
  for (int i = tid; i < HC_WIH2; i += NTHREADS) sWih2[i] = (h16)Wih2[i];
  for (int i = tid; i < HC_WHH2; i += NTHREADS) sWhh2[i] = (h16)Whh2[i];
  for (int i = tid; i < HC_WIH3; i += NTHREADS) sWih3[i] = (h16)Wih3[i];
  for (int i = tid; i < HC_WHH3; i += NTHREADS) sWhh3[i] = (h16)Whh3[i];
  for (int i = tid; i < FB1; i += NTHREADS) sB1[i] = bih1[i] + bhh1[i];
  for (int i = tid; i < FB2; i += NTHREADS) sB2[i] = bih2[i] + bhh2[i];
  for (int i = tid; i < FB3; i += NTHREADS) sB3[i] = bih3[i] + bhh3[i];
  for (int i = tid; i < HC_H1; i += NTHREADS) sH1[i] = (h16)0.f;
  for (int i = tid; i < HC_H2; i += NTHREADS) sH2[i] = (h16)0.f;
  for (int i = tid; i < HC_H3; i += NTHREADS) sH3[i] = (h16)0.f;

  LSTMTile<IN_, H1_> t1; t1.init();   // all 8 waves: hidden slice [16*wave, +16) of 128
  LSTMTile<H1_, H2_> t2; t2.init();   // waves 0..3:  slice of 64
  LSTMTile<H2_, H3_> t3; t3.init();   // waves 0..1:  slice of 32

  __syncthreads();

  int pp = 0;  // ping-pong: read h(t-1) from [pp], write h(t) to [pp^1]
  for (int t = 0; t < T_; ++t) {
    // stage x_t tile (16 rows x 32) as f16
    for (int i = tid; i < HC_XT; i += NTHREADS) {
      const int r = i >> 5, k = i & 31;
      sXt[i] = (h16)x[((size_t)(row0 + r) * T_ + t) * IN_ + k];
    }
    __syncthreads();   // x_t visible; also separates prior-step reads from this-step writes

    // ---- layer 1 (all waves): read sH1[pp], write sH1[pp^1] ----
    t1.step<false>(sXt, sH1 + pp * HC_H1, sH1 + (pp ^ 1) * HC_H1,
                   sWih1, sWhh1, sB1, wave * 16, lane);
    __syncthreads();   // h1(t) visible

    // ---- layer 2 (waves 0..3), input = relu(h1(t)) ----
    if (wave < 4)
      t2.step<true>(sH1 + (pp ^ 1) * HC_H1, sH2 + pp * HC_H2, sH2 + (pp ^ 1) * HC_H2,
                    sWih2, sWhh2, sB2, wave * 16, lane);
    __syncthreads();   // h2(t) visible

    // ---- layer 3 (waves 0..1), input = relu(h2(t)) ----
    if (wave < 2)
      t3.step<true>(sH2 + (pp ^ 1) * HC_H2, sH3 + pp * HC_H3, sH3 + (pp ^ 1) * HC_H3,
                    sWih3, sWhh3, sB3, wave * 16, lane);
    // no barrier: next-iteration barriers order all uses of sH3[pp^1]

    pp ^= 1;
  }
  __syncthreads();   // h3(T-1) visible for the dense head

  // ---- dense head on relu(h3) at t = T-1 (final buffer index = pp after loop) ----
  const h16* sH3fin = sH3 + pp * HC_H3;
  if (tid < RPB) {
    float acc = bd[0];
#pragma unroll
    for (int k = 0; k < H3_; ++k) {
      float hv = (float)sH3fin[tid * H3_ + k];
      hv = fmaxf(hv, 0.0f);
      acc += hv * Wd[k];
    }
    out[row0 + tid] = acc;
  }
}

extern "C" void kernel_launch(void* const* d_in, const int* in_sizes, int n_in,
                              void* d_out, int out_size, void* d_ws, size_t ws_size,
                              hipStream_t stream) {
  (void)in_sizes; (void)n_in; (void)out_size; (void)d_ws; (void)ws_size;
  const float* x    = (const float*)d_in[0];
  const float* Wih1 = (const float*)d_in[1];
  const float* Whh1 = (const float*)d_in[2];
  const float* bih1 = (const float*)d_in[3];
  const float* bhh1 = (const float*)d_in[4];
  const float* Wih2 = (const float*)d_in[5];
  const float* Whh2 = (const float*)d_in[6];
  const float* bih2 = (const float*)d_in[7];
  const float* bhh2 = (const float*)d_in[8];
  const float* Wih3 = (const float*)d_in[9];
  const float* Whh3 = (const float*)d_in[10];
  const float* bih3 = (const float*)d_in[11];
  const float* bhh3 = (const float*)d_in[12];
  const float* Wd   = (const float*)d_in[13];
  const float* bd   = (const float*)d_in[14];

  hipFuncSetAttribute(reinterpret_cast<const void*>(lstm3_fused_kernel),
                      hipFuncAttributeMaxDynamicSharedMemorySize, (int)LDS_BYTES);

  lstm3_fused_kernel<<<B_ / RPB, NTHREADS, LDS_BYTES, stream>>>(
      x, Wih1, Whh1, bih1, bhh1, Wih2, Whh2, bih2, bhh2,
      Wih3, Whh3, bih3, bhh3, Wd, bd, (float*)d_out);
}
